// GVAE_RGCN_64046552318137
// MI455X (gfx1250) — matrix operand order repaired and checked by reading, not compiled, not run on hardware
//
#include <hip/hip_runtime.h>
#include <hip/hip_bf16.h>

typedef __bf16 bf16_t;
typedef __attribute__((ext_vector_type(16))) __bf16 v16bf;
typedef __attribute__((ext_vector_type(8)))  float  v8f;

#define WMMA_BF16(A, Bm, C) \
  __builtin_amdgcn_wmma_f32_16x16x32_bf16(false, (A), false, (Bm), (short)0, (C), false, false)

namespace {
constexpr int N    = 65536;
constexpr int BG   = 512;
constexpr int NPG  = 128;
constexpr int E    = 524288;
constexpr int EPG  = 1024;
constexpr int D    = 64;
constexpr int R    = 16;
constexpr int L    = 3;
constexpr int LD   = L * D;           // 192
constexpr int PTOT  = E + 16 * R;     // padded perm length 524544
constexpr int TILES = PTOT / 16;      // 32784 edge tiles
constexpr int MSG_TPW = 8;            // edge tiles per wave
constexpr int UPD_TPW = 4;            // row tiles per wave
}

__device__ __forceinline__ float atomAddF(float* p, float v) {
  return __hip_atomic_fetch_add(p, v, __ATOMIC_RELAXED, __HIP_MEMORY_SCOPE_AGENT);
}
__device__ __forceinline__ float bflo(unsigned u) { return __uint_as_float(u << 16); }
__device__ __forceinline__ float bfhi(unsigned u) { return __uint_as_float(u & 0xFFFF0000u); }

// ---- WMMA fragment loaders (ISA 7.12.2 layouts, wave32) ----
union AFrag { v16bf v; uint4 q[2]; };

__device__ __forceinline__ v16bf loadA(const bf16_t* __restrict__ row, int kb, int hi) {
  AFrag a;
  a.q[0] = *(const uint4*)(row + kb + hi * 8);
  a.q[1] = *(const uint4*)(row + kb + 16 + hi * 8);
  return a.v;
}

struct BW { v16bf f[4][2]; };  // [n-tile][k-tile] for a 64x64 weight

__device__ __forceinline__ void loadB(const bf16_t* __restrict__ W, BW& b) {
  int lane = threadIdx.x & 31, i = lane & 15, hi = lane >> 4;
#pragma unroll
  for (int nb = 0; nb < 4; ++nb)
#pragma unroll
    for (int kt = 0; kt < 2; ++kt) {
      union { v16bf v; bf16_t e[16]; } u;
      int kbase = kt * 32 + hi * 8;
#pragma unroll
      for (int j = 0; j < 16; ++j) u.e[j] = W[(kbase + j + (j >= 8 ? 8 : 0)) * 64 + nb * 16 + i];
      b.f[nb][kt] = u.v;
    }
}

// ---- utility kernels ----
__global__ void k_fill_f32(float* p, int n, float v) {
  int i = blockIdx.x * blockDim.x + threadIdx.x;
  if (i < n) p[i] = v;
}
__global__ void k_fill_i32(int* p, int n, int v) {
  int i = blockIdx.x * blockDim.x + threadIdx.x;
  if (i < n) p[i] = v;
}
__global__ void k_f32_to_bf16(const float* __restrict__ in, bf16_t* __restrict__ out, int n) {
  int i = blockIdx.x * blockDim.x + threadIdx.x;
  if (i < n) out[i] = (bf16_t)in[i];
}

// tiny dense: out[r,k] = relu(in[r,:] @ W + b)   (R x 64)
__global__ void k_relenc(const float* __restrict__ in, const float* __restrict__ W,
                         const float* __restrict__ b, float* __restrict__ out) {
  int r = blockIdx.x, k = threadIdx.x;
  float acc = b[k];
  for (int d2 = 0; d2 < 64; ++d2) acc += in[r * 64 + d2] * W[d2 * 64 + k];
  out[r * 64 + k] = fmaxf(acc, 0.f);
}

// ---- counting sort of edges by relation (LDS-privatized histogram) ----
__global__ void k_hist(const int* __restrict__ et, const int* __restrict__ aet,
                       int* cnto, int* cnta) {
  __shared__ int lh[2 * R];
  if (threadIdx.x < 2 * R) lh[threadIdx.x] = 0;
  __syncthreads();
  int e = blockIdx.x * blockDim.x + threadIdx.x;
  if (e < E) { atomicAdd(&lh[et[e]], 1); atomicAdd(&lh[R + aet[e]], 1); }
  __syncthreads();
  if (threadIdx.x < R) atomicAdd(cnto + threadIdx.x, lh[threadIdx.x]);
  else if (threadIdx.x < 2 * R) atomicAdd(cnta + threadIdx.x - R, lh[threadIdx.x]);
}
__global__ void k_seg(const int* cnto, const int* cnta, int* sego, int* sega) {
  if (threadIdx.x == 0 && blockIdx.x == 0) {
    int so = 0, sa = 0;
    for (int r = 0; r < R; ++r) {
      sego[r] = so; so += ((cnto[r] + 15) >> 4) << 4;
      sega[r] = sa; sa += ((cnta[r] + 15) >> 4) << 4;
    }
    sego[R] = so; sega[R] = sa;
  }
}
// block-level reservation: 16 global atomics per block instead of 1 per edge
__global__ void k_scatter(const int* __restrict__ et, const int* __restrict__ seg,
                          int* cur, int* perm) {
  __shared__ int lh[R], lbase[R];
  if (threadIdx.x < R) lh[threadIdx.x] = 0;
  __syncthreads();
  int e = blockIdx.x * blockDim.x + threadIdx.x;
  int r = 0, mypos = 0;
  if (e < E) { r = et[e]; mypos = atomicAdd(&lh[r], 1); }
  __syncthreads();
  if (threadIdx.x < R) lbase[threadIdx.x] = atomicAdd(cur + threadIdx.x, lh[threadIdx.x]);
  __syncthreads();
  if (e < E) perm[seg[r] + lbase[r] + mypos] = e;
}

// ---- RGCN message pass: WMMA over 16-edge tiles (sorted by relation) ----
// Edge weights are exactly {0,1} (ones or thresholded mask), so validity is a
// wave-uniform ballot bitmask; no per-element float multiply/compare needed.
__global__ void k_msg(const bf16_t* __restrict__ h, const bf16_t* __restrict__ wrel_l,
                      const int* __restrict__ perm, const int* __restrict__ seg,
                      const int* __restrict__ src, const int* __restrict__ dst,
                      const float* __restrict__ wmask, float* __restrict__ agg,
                      float* __restrict__ deg) {
  int wid = (blockIdx.x * blockDim.x + threadIdx.x) >> 5;
  int lane = threadIdx.x & 31, i = lane & 15, hi = lane >> 4;
  int t0 = wid * MSG_TPW;
  // seg boundaries are wave-uniform: force into SGPRs to relieve VGPR pressure
  int segr[R];
#pragma unroll
  for (int q = 0; q < R; ++q) segr[q] = __builtin_amdgcn_readfirstlane(seg[q]);
  BW bw; int cur_r = -1;
  for (int tt = 0; tt < MSG_TPW; ++tt) {
    int t = t0 + tt;
    if (t >= TILES) return;
    if (t + 1 < TILES) __builtin_prefetch(perm + (t + 1) * 16 + i, 0, 1);
    int e = perm[t * 16 + i];
    bool val = (e >= 0);
    if (val && wmask) val = (wmask[e] != 0.f);
    unsigned vm = (unsigned)__ballot(val);        // bits 0-15 = rows 0-15 (16-31 dup)
    if (vm == 0u) continue;                       // wave-uniform skip
    int base = t * 16, r = 0;
#pragma unroll
    for (int q = 1; q < R; ++q) r += (segr[q] <= base) ? 1 : 0;
    r = __builtin_amdgcn_readfirstlane(r);        // scalar relation index
    if (r != cur_r) { loadB(wrel_l + r * 4096, bw); cur_r = r; }
    int sN = (e >= 0) ? src[e] : 0;
    int dN = val ? dst[e] : 0;
    const bf16_t* hrow = h + sN * 64;
    v16bf a0 = loadA(hrow, 0, hi);
    v16bf a1 = loadA(hrow, 32, hi);
    if (hi == 0 && val) atomAddF(deg + dN, 1.0f);
    // compute all four 16x16 column tiles first (full 16x64 message tile)
    v8f cacc[4];
#pragma unroll
    for (int nb = 0; nb < 4; ++nb) {
      v8f c = {0.f, 0.f, 0.f, 0.f, 0.f, 0.f, 0.f, 0.f};
      c = WMMA_BF16(a0, bw.f[nb][0], c);
      cacc[nb] = WMMA_BF16(a1, bw.f[nb][1], c);
    }
    // hoisted row broadcasts: 8 bpermutes/tile; j-outer scatter so each row's
    // predicate covers 4 atomics at constant offsets from one base pointer
#pragma unroll
    for (int j = 0; j < 8; ++j) {
      unsigned dr = (unsigned)__shfl(dN, j + hi * 8);
      float* p = agg + (size_t)dr * 64 + i;
      if ((vm >> (j + hi * 8)) & 1u) {
        atomAddF(p,      cacc[0][j]);
        atomAddF(p + 16, cacc[1][j]);
        atomAddF(p + 32, cacc[2][j]);
        atomAddF(p + 48, cacc[3][j]);
      }
    }
  }
}

// ---- node update: h = relu(h@Wself + agg/max(deg,1) + b); pool; store reps ----
__global__ void k_upd(const bf16_t* __restrict__ wself_l, const float* __restrict__ bias_l,
                      bf16_t* __restrict__ h, const float* __restrict__ agg,
                      const float* __restrict__ deg, float* __restrict__ rep, int l,
                      float* __restrict__ xout, float* __restrict__ pool) {
  int wid = (blockIdx.x * blockDim.x + threadIdx.x) >> 5;
  int lane = threadIdx.x & 31, i = lane & 15, hi = lane >> 4;
  BW bw; loadB(wself_l, bw);
  float bv[4];
#pragma unroll
  for (int nb = 0; nb < 4; ++nb) bv[nb] = bias_l[nb * 16 + i];
  int t0 = wid * UPD_TPW;
  for (int tt = 0; tt < UPD_TPW; ++tt) {
    int t = t0 + tt;
    if (t >= N / 16) return;
    int m0 = t * 16;
    const bf16_t* hrow = h + (m0 + i) * 64;
    v16bf a0 = loadA(hrow, 0, hi);
    v16bf a1 = loadA(hrow, 32, hi);
    v8f cacc[4];
#pragma unroll
    for (int nb = 0; nb < 4; ++nb) {
      v8f c = {0.f, 0.f, 0.f, 0.f, 0.f, 0.f, 0.f, 0.f};
      c = WMMA_BF16(a0, bw.f[nb][0], c);
      cacc[nb] = WMMA_BF16(a1, bw.f[nb][1], c);
    }
    float psum[4] = {0.f, 0.f, 0.f, 0.f};
#pragma unroll
    for (int j = 0; j < 8; ++j) {
      int n = m0 + j + hi * 8;
      float inv = 1.0f / fmaxf(deg[n], 1.0f);
#pragma unroll
      for (int nb = 0; nb < 4; ++nb) {
        int col = nb * 16 + i;
        float val = fmaxf(cacc[nb][j] + agg[n * 64 + col] * inv + bv[nb], 0.f);
        h[n * 64 + col] = (bf16_t)val;
        if (rep)  rep[n * LD + l * 64 + col] = val;
        if (xout) xout[n * 64 + col] = val;
        psum[nb] += val;
      }
    }
    // all 16 rows of a tile belong to one graph (NPG=128, 16-aligned)
#pragma unroll
    for (int nb = 0; nb < 4; ++nb)
      atomAddF(pool + (m0 >> 7) * LD + l * 64 + nb * 16 + i, psum[nb]);
  }
}

// ---- AdaIN style transfer: per-row stats (ddof=1) ----
__global__ void k_style(const float* __restrict__ repg, const float* __restrict__ augx,
                        bf16_t* __restrict__ style) {
  int wid = (blockIdx.x * blockDim.x + threadIdx.x) >> 5;
  int lane = threadIdx.x & 31;
  if (wid >= N) return;
  const float* x  = repg + wid * LD + 2 * 64;   // layer-2 slice = x
  const float* ax = augx + wid * 64;
  float x0 = x[lane], x1 = x[lane + 32];
  float a0 = ax[lane], a1 = ax[lane + 32];
  float s = x0 + x1, ss = x0 * x0 + x1 * x1;
  float sa = a0 + a1, ssa = a0 * a0 + a1 * a1;
#pragma unroll
  for (int o = 16; o >= 1; o >>= 1) {
    s += __shfl_xor(s, o); ss += __shfl_xor(ss, o);
    sa += __shfl_xor(sa, o); ssa += __shfl_xor(ssa, o);
  }
  float mc = s / 64.f, ms = sa / 64.f;
  float sc = sqrtf((ss - 64.f * mc * mc) / 63.f + 1e-5f);
  float sv = sqrtf((ssa - 64.f * ms * ms) / 63.f + 1e-5f);
  style[wid * 64 + lane]      = (bf16_t)(sv * ((x0 - mc) / sc) + ms);
  style[wid * 64 + lane + 32] = (bf16_t)(sv * ((x1 - mc) / sc) + ms);
}

// ---- VAE reparam: z = style@Wm + sqrt(exp(style@Ws)) * eps ----
__global__ void k_vae(const bf16_t* __restrict__ style, const bf16_t* __restrict__ wm,
                      const bf16_t* __restrict__ wsd, const float* __restrict__ eps,
                      bf16_t* __restrict__ z) {
  int wid = (blockIdx.x * blockDim.x + threadIdx.x) >> 5;
  int lane = threadIdx.x & 31, i = lane & 15, hi = lane >> 4;
  BW bm, bs; loadB(wm, bm); loadB(wsd, bs);
  int t0 = wid * UPD_TPW;
  for (int tt = 0; tt < UPD_TPW; ++tt) {
    int t = t0 + tt;
    if (t >= N / 16) return;
    int m0 = t * 16;
    const bf16_t* srow = style + (m0 + i) * 64;
    v16bf a0 = loadA(srow, 0, hi), a1 = loadA(srow, 32, hi);
#pragma unroll
    for (int nb = 0; nb < 4; ++nb) {
      v8f cm = {0.f,0.f,0.f,0.f,0.f,0.f,0.f,0.f};
      v8f cs = {0.f,0.f,0.f,0.f,0.f,0.f,0.f,0.f};
      cm = WMMA_BF16(a0, bm.f[nb][0], cm); cm = WMMA_BF16(a1, bm.f[nb][1], cm);
      cs = WMMA_BF16(a0, bs.f[nb][0], cs); cs = WMMA_BF16(a1, bs.f[nb][1], cs);
      int col = nb * 16 + i;
#pragma unroll
      for (int j = 0; j < 8; ++j) {
        int n = m0 + j + hi * 8;
        z[n * 64 + col] = (bf16_t)(cm[j] + sqrtf(expf(cs[j])) * eps[n * 64 + col]);
      }
    }
  }
}

// ---- decoder node transforms: zh = relu(z@Wh+bh), zt = relu(z@Wt+bt) ----
__global__ void k_dec(const bf16_t* __restrict__ z, const bf16_t* __restrict__ wh,
                      const bf16_t* __restrict__ wt, const float* __restrict__ bh,
                      const float* __restrict__ bt, bf16_t* __restrict__ zh,
                      bf16_t* __restrict__ zt) {
  int wid = (blockIdx.x * blockDim.x + threadIdx.x) >> 5;
  int lane = threadIdx.x & 31, i = lane & 15, hi = lane >> 4;
  BW bhW, btW; loadB(wh, bhW); loadB(wt, btW);
  int t0 = wid * UPD_TPW;
  for (int tt = 0; tt < UPD_TPW; ++tt) {
    int t = t0 + tt;
    if (t >= N / 16) return;
    int m0 = t * 16;
    const bf16_t* zrow = z + (m0 + i) * 64;
    v16bf a0 = loadA(zrow, 0, hi), a1 = loadA(zrow, 32, hi);
#pragma unroll
    for (int nb = 0; nb < 4; ++nb) {
      v8f ch = {0.f,0.f,0.f,0.f,0.f,0.f,0.f,0.f};
      v8f ct = {0.f,0.f,0.f,0.f,0.f,0.f,0.f,0.f};
      ch = WMMA_BF16(a0, bhW.f[nb][0], ch); ch = WMMA_BF16(a1, bhW.f[nb][1], ch);
      ct = WMMA_BF16(a0, btW.f[nb][0], ct); ct = WMMA_BF16(a1, btW.f[nb][1], ct);
      int col = nb * 16 + i;
      float bhv = bh[col], btv = bt[col];
#pragma unroll
      for (int j = 0; j < 8; ++j) {
        int n = m0 + j + hi * 8;
        zh[n * 64 + col] = (bf16_t)fmaxf(ch[j] + bhv, 0.f);
        zt[n * 64 + col] = (bf16_t)fmaxf(ct[j] + btv, 0.f);
      }
    }
  }
}

// ---- DistMult edge scoring -> mask (vectorized b128 loads, bf16 bit-unpack) ----
__global__ void k_mask(const bf16_t* __restrict__ zh, const bf16_t* __restrict__ zt,
                       const float* __restrict__ reld, const int* __restrict__ src,
                       const int* __restrict__ dst, const int* __restrict__ et,
                       float* __restrict__ mask) {
  int e = blockIdx.x * blockDim.x + threadIdx.x;
  if (e >= E) return;
  const uint4*  ph = (const uint4*)(zh + src[e] * 64);
  const uint4*  pt = (const uint4*)(zt + dst[e] * 64);
  const float4* pr = (const float4*)(reld + et[e] * 64);
  float acc = 0.f;
#pragma unroll
  for (int q = 0; q < 8; ++q) {
    uint4  hq = ph[q], tq = pt[q];
    float4 ra = pr[2 * q], rb = pr[2 * q + 1];
    acc += bflo(hq.x) * ra.x * bflo(tq.x) + bfhi(hq.x) * ra.y * bfhi(tq.x)
         + bflo(hq.y) * ra.z * bflo(tq.y) + bfhi(hq.y) * ra.w * bfhi(tq.y)
         + bflo(hq.z) * rb.x * bflo(tq.z) + bfhi(hq.z) * rb.y * bfhi(tq.z)
         + bflo(hq.w) * rb.z * bflo(tq.w) + bfhi(hq.w) * rb.w * bfhi(tq.w);
  }
  float m = (acc >= 0.5f) ? 1.f : 0.f;
  if ((e & (EPG - 1)) == 0) m = 1.f;   // root edge forced kept
  mask[e] = m;
}

// ---- loss: per-(n,l) row -> mse + style stats diffs ----
__global__ void k_loss(const float* __restrict__ rec, const float* __restrict__ rg,
                       float* __restrict__ acc) {
  int wid = (blockIdx.x * blockDim.x + threadIdx.x) >> 5;
  int lane = threadIdx.x & 31;
  if (wid >= N * L) return;
  const float* a = rec + wid * 64;
  const float* b = rg + wid * 64;
  float a0 = a[lane], a1 = a[lane + 32], b0 = b[lane], b1 = b[lane + 32];
  float d0 = a0 - b0, d1 = a1 - b1;
  float sd = d0 * d0 + d1 * d1;
  float sa = a0 + a1, ssa = a0 * a0 + a1 * a1;
  float sb = b0 + b1, ssb = b0 * b0 + b1 * b1;
#pragma unroll
  for (int o = 16; o >= 1; o >>= 1) {
    sd += __shfl_xor(sd, o);
    sa += __shfl_xor(sa, o); ssa += __shfl_xor(ssa, o);
    sb += __shfl_xor(sb, o); ssb += __shfl_xor(ssb, o);
  }
  if (lane == 0) {
    atomAddF(acc + 0, sd);
    float ma = sa / 64.f, mb = sb / 64.f;
    float da = sqrtf((ssa - 64.f * ma * ma) / 63.f + 1e-5f);
    float db = sqrtf((ssb - 64.f * mb * mb) / 63.f + 1e-5f);
    atomAddF(acc + 1, (ma - mb) * (ma - mb) + (da - db) * (da - db));
  }
}

// ---- final FC: g_rep (640) @ fc_W + fc_b, one wave per graph ----
__global__ void k_fc(const float* __restrict__ poolr, const float* __restrict__ repr,
                     const float* __restrict__ embrel, const int* __restrict__ rlab,
                     const float* __restrict__ fcW, const float* __restrict__ fcb,
                     float* __restrict__ out) {
  int b = (blockIdx.x * blockDim.x + threadIdx.x) >> 5;
  int lane = threadIdx.x & 31;
  if (b >= BG) return;
  const float* hrow = repr + (b * NPG) * LD;       // head node (id 0 in graph)
  const float* trow = repr + (b * NPG + 1) * LD;   // tail node
  const float* prow = poolr + b * LD;
  const float* erow = embrel + rlab[b] * 64;
  float acc = 0.f;
  for (int k = lane; k < 3 * LD + 64; k += 32) {
    float v;
    if (k < LD)            v = prow[k] * (1.f / NPG);
    else if (k < 2 * LD)   v = hrow[k - LD];
    else if (k < 3 * LD)   v = trow[k - 2 * LD];
    else                   v = erow[k - 3 * LD];
    acc += v * fcW[k];
  }
#pragma unroll
  for (int o = 16; o >= 1; o >>= 1) acc += __shfl_xor(acc, o);
  if (lane == 0) out[b] = acc + fcb[0];
}

__global__ void k_final(const float* __restrict__ pg, const float* __restrict__ pa,
                        const float* __restrict__ acc, float* __restrict__ out) {
  int i = blockIdx.x * blockDim.x + threadIdx.x;
  if (i < BG * LD) {
    out[BG + i]           = pg[i] * (1.f / NPG);
    out[BG + BG * LD + i] = pa[i] * (1.f / NPG);
  }
  if (i == 0)
    out[BG + 2 * BG * LD] = acc[0] / (float)(N * 64) + 0.1f * (acc[1] / (float)N);
}

extern "C" void kernel_launch(void* const* d_in, const int* in_sizes, int n_in,
                              void* d_out, int out_size, void* d_ws, size_t ws_size,
                              hipStream_t stream) {
  (void)in_sizes; (void)n_in; (void)out_size; (void)ws_size;
  const float* feat   = (const float*)d_in[0];
  const float* relft  = (const float*)d_in[1];
  const float* rpW    = (const float*)d_in[2];
  const float* rpB    = (const float*)d_in[3];
  const float* Wrel   = (const float*)d_in[4];
  const float* Wself  = (const float*)d_in[5];
  const float* rb     = (const float*)d_in[6];
  const float* Wmean  = (const float*)d_in[7];
  const float* Wstd   = (const float*)d_in[8];
  const float* dWh    = (const float*)d_in[9];
  const float* dbh    = (const float*)d_in[10];
  const float* dWt    = (const float*)d_in[11];
  const float* dbt    = (const float*)d_in[12];
  const float* dWr    = (const float*)d_in[13];
  const float* dbr    = (const float*)d_in[14];
  const float* fcW    = (const float*)d_in[15];
  const float* fcb    = (const float*)d_in[16];
  const float* eps    = (const float*)d_in[17];
  const int*   src    = (const int*)d_in[18];
  const int*   dst    = (const int*)d_in[19];
  const int*   et     = (const int*)d_in[20];
  const int*   asrc   = (const int*)d_in[21];
  const int*   adst   = (const int*)d_in[22];
  const int*   aet    = (const int*)d_in[23];
  const int*   rlab   = (const int*)d_in[25];
  float* out = (float*)d_out;

  // ---- workspace carve-up ----
  char* ws = (char*)d_ws;
  size_t off = 0;
  auto alloc = [&](size_t bytes) -> void* {
    void* p = ws + off;
    off += (bytes + 255) & ~(size_t)255;
    return p;
  };
  bf16_t* wrel_b  = (bf16_t*)alloc((size_t)L * R * 4096 * 2);
  bf16_t* wself_b = (bf16_t*)alloc((size_t)L * 4096 * 2);
  bf16_t* wmean_b = (bf16_t*)alloc(4096 * 2);
  bf16_t* wstd_b  = (bf16_t*)alloc(4096 * 2);
  bf16_t* wh_b    = (bf16_t*)alloc(4096 * 2);
  bf16_t* wt_b    = (bf16_t*)alloc(4096 * 2);
  float*  embrel  = (float*)alloc(R * 64 * 4);
  float*  reld    = (float*)alloc(R * 64 * 4);
  bf16_t* hbuf    = (bf16_t*)alloc((size_t)N * 64 * 2);
  float*  agg     = (float*)alloc((size_t)N * 64 * 4);
  float*  deg     = (float*)alloc((size_t)N * 4);
  float*  rep_g   = (float*)alloc((size_t)N * LD * 4);
  float*  rep_r   = (float*)alloc((size_t)N * LD * 4);
  float*  augx    = (float*)alloc((size_t)N * 64 * 4);
  bf16_t* styleb  = (bf16_t*)alloc((size_t)N * 64 * 2);
  bf16_t* zbuf    = (bf16_t*)alloc((size_t)N * 64 * 2);
  bf16_t* zhb     = (bf16_t*)alloc((size_t)N * 64 * 2);
  bf16_t* ztb     = (bf16_t*)alloc((size_t)N * 64 * 2);
  float*  maskb   = (float*)alloc((size_t)E * 4);
  int*    perm_o  = (int*)alloc((size_t)PTOT * 4);
  int*    perm_a  = (int*)alloc((size_t)PTOT * 4);
  int*    ibase   = (int*)alloc(128 * 4);
  float*  pools   = (float*)alloc((size_t)3 * BG * LD * 4);
  float*  lacc    = (float*)alloc(2 * 4);
  int *cnto = ibase + 0,  *cnta = ibase + 16;
  int *curo = ibase + 32, *cura = ibase + 48;
  int *sego = ibase + 64, *sega = ibase + 84;
  float *pool_g = pools, *pool_a = pools + BG * LD, *pool_r = pools + 2 * BG * LD;

  auto cdiv = [](int a, int b) { return (a + b - 1) / b; };

  // ---- zero / init (every call: graph-replay safe) ----
  k_fill_i32<<<cdiv(PTOT, 256), 256, 0, stream>>>(perm_o, PTOT, -1);
  k_fill_i32<<<cdiv(PTOT, 256), 256, 0, stream>>>(perm_a, PTOT, -1);
  k_fill_i32<<<1, 64, 0, stream>>>(ibase, 64, 0);
  k_fill_f32<<<cdiv(3 * BG * LD, 256), 256, 0, stream>>>(pools, 3 * BG * LD, 0.f);
  k_fill_f32<<<1, 2, 0, stream>>>(lacc, 2, 0.f);

  // ---- weight conversion to bf16 ----
  k_f32_to_bf16<<<cdiv(L * R * 4096, 256), 256, 0, stream>>>(Wrel, wrel_b, L * R * 4096);
  k_f32_to_bf16<<<cdiv(L * 4096, 256), 256, 0, stream>>>(Wself, wself_b, L * 4096);
  k_f32_to_bf16<<<16, 256, 0, stream>>>(Wmean, wmean_b, 4096);
  k_f32_to_bf16<<<16, 256, 0, stream>>>(Wstd, wstd_b, 4096);
  k_f32_to_bf16<<<16, 256, 0, stream>>>(dWh, wh_b, 4096);
  k_f32_to_bf16<<<16, 256, 0, stream>>>(dWt, wt_b, 4096);

  // ---- relation encoder + decoder relation transform ----
  k_relenc<<<R, 64, 0, stream>>>(relft, rpW, rpB, embrel);
  k_relenc<<<R, 64, 0, stream>>>(embrel, dWr, dbr, reld);

  // ---- counting sort of both edge sets by relation ----
  k_hist<<<cdiv(E, 256), 256, 0, stream>>>(et, aet, cnto, cnta);
  k_seg<<<1, 1, 0, stream>>>(cnto, cnta, sego, sega);
  k_scatter<<<cdiv(E, 256), 256, 0, stream>>>(et, sego, curo, perm_o);
  k_scatter<<<cdiv(E, 256), 256, 0, stream>>>(aet, sega, cura, perm_a);

  const int MSG_BLOCKS = cdiv(TILES, MSG_TPW * 4);   // 4 waves/block
  const int UPD_BLOCKS = cdiv(N / 16, UPD_TPW * 4);

  auto encode = [&](const int* perm, const int* seg, const int* esrc, const int* edst,
                    const float* wm, float* rep, float* xout_last, float* pool) {
    k_f32_to_bf16<<<cdiv(N * 64, 256), 256, 0, stream>>>(feat, hbuf, N * 64);
    for (int l = 0; l < L; ++l) {
      k_fill_f32<<<cdiv(N * 64, 256), 256, 0, stream>>>(agg, N * 64, 0.f);
      k_fill_f32<<<cdiv(N, 256), 256, 0, stream>>>(deg, N, 0.f);
      k_msg<<<MSG_BLOCKS, 128, 0, stream>>>(hbuf, wrel_b + (size_t)l * R * 4096, perm, seg,
                                            esrc, edst, wm, agg, deg);
      k_upd<<<UPD_BLOCKS, 128, 0, stream>>>(wself_b + (size_t)l * 4096, rb + l * 64, hbuf,
                                            agg, deg, rep, l,
                                            (l == L - 1) ? xout_last : nullptr, pool);
    }
  };

  // original / augmented encodes
  encode(perm_o, sego, src, dst, nullptr, rep_g, nullptr, pool_g);
  encode(perm_a, sega, asrc, adst, nullptr, nullptr, augx, pool_a);

  // AdaIN + VAE + decoder
  k_style<<<cdiv(N * 32, 256), 256, 0, stream>>>(rep_g, augx, styleb);
  k_vae<<<UPD_BLOCKS, 128, 0, stream>>>(styleb, wmean_b, wstd_b, eps, zbuf);
  k_dec<<<UPD_BLOCKS, 128, 0, stream>>>(zbuf, wh_b, wt_b, dbh, dbt, zhb, ztb);
  k_mask<<<cdiv(E, 256), 256, 0, stream>>>(zhb, ztb, reld, src, dst, et, maskb);

  // reconstructed-graph encode (masked weights, same topology/perm)
  encode(perm_o, sego, src, dst, maskb, rep_r, nullptr, pool_r);

  // losses + outputs
  k_loss<<<cdiv(N * L * 32, 256), 256, 0, stream>>>(rep_r, rep_g, lacc);
  k_fc<<<cdiv(BG * 32, 256), 256, 0, stream>>>(pool_r, rep_r, embrel, rlab, fcW, fcb, out);
  k_final<<<cdiv(BG * LD, 256), 256, 0, stream>>>(pool_g, pool_a, lacc, out);
}
